// MaskSeparator_35270271435342
// MI455X (gfx1250) — matrix operand order repaired and checked by reading, not compiled, not run on hardware
//
#include <hip/hip_runtime.h>
#include <hip/hip_bf16.h>
#include <math.h>

// ---------------------------------------------------------------------------
// MaskSeparator forward for MI455X (gfx1250, wave32, WMMA).
//   pre GEMM+LN -> 2x LSTM distributed over 16 WGPs (bf16 WMMA, weight shard
//   pinned in 296KB LDS, c-state in registers, h exchanged via L2 + grid
//   barrier, A-tile staged with global_load_async_to_lds) -> post LN ->
//   routing top-2 -> 3x head GEMM + ReLU/LN/gate.
// ---------------------------------------------------------------------------

#define BB 16
#define TT 1000
#define DD 512
#define HH 512
#define NHEAD 3
#define NLAYER 2
#define ROWS (BB * TT)   // 16000
#define NWG  16          // workgroups for the distributed LSTM

typedef __attribute__((ext_vector_type(16))) __bf16        v16bf;
typedef __attribute__((ext_vector_type(8)))  float         v8f;
typedef __attribute__((ext_vector_type(4)))  unsigned int  u32x4;

union FragAB {
    v16bf        v;
    unsigned int u[8];
    u32x4        q[2];
};

__device__ __forceinline__ unsigned short f2bf(float f) {
    unsigned int u = __float_as_uint(f);
    u += 0x7FFFu + ((u >> 16) & 1u);       // round-to-nearest-even
    return (unsigned short)(u >> 16);
}

__device__ __forceinline__ float sigm(float x) { return 1.0f / (1.0f + __expf(-x)); }

// CDNA5 async global->LDS DMA (ASYNCcnt-tracked); lds_byte_off is the byte
// offset within the workgroup's LDS allocation (dynamic LDS starts at 0).
__device__ __forceinline__ void async_g2l_b128(unsigned int lds_byte_off, const void* gptr) {
    asm volatile("global_load_async_to_lds_b128 %0, %1, off"
                 :
                 : "v"(lds_byte_off), "v"(gptr)
                 : "memory");
}
__device__ __forceinline__ void wait_asynccnt0() {
    asm volatile("s_wait_asynccnt 0" ::: "memory");
}

// ---------------------------------------------------------------------------
__global__ void k_f32_to_bf16(const float* __restrict__ in,
                              unsigned short* __restrict__ out, int n) {
    int i = blockIdx.x * blockDim.x + threadIdx.x;
    if (i < n) out[i] = f2bf(in[i]);
}

__global__ void k_zero32(unsigned int* __restrict__ p, int n) {
    int i = blockIdx.x * blockDim.x + threadIdx.x;
    if (i < n) p[i] = 0u;
}

// ---------------------------------------------------------------------------
// C[M,N] = A(MxK, bf16 row-major) * W(NxK, bf16 row-major)^T + bias.
// One wave per 16x16 output tile; K-loop of V_WMMA_F32_16X16X32_BF16.
__global__ void k_wmma_gemm_bias(const unsigned short* __restrict__ A,
                                 const unsigned short* __restrict__ W,
                                 const float* __restrict__ bias,
                                 float* __restrict__ C,
                                 int M, int N, int K) {
    const int lane  = threadIdx.x & 31;
    const int wave  = (blockIdx.x * blockDim.x + threadIdx.x) >> 5;
    const int ntile = N >> 4;
    const int mt    = wave / ntile;
    const int nt    = wave - mt * ntile;
    if (mt >= (M >> 4)) return;            // wave-uniform: EXEC stays all-ones
    const int half = lane >> 4;
    const int l15  = lane & 15;

    const unsigned int*   arow = (const unsigned int*)(A + (size_t)(mt * 16 + l15) * K);
    const unsigned short* wrow = W + (size_t)(nt * 16 + l15) * K;

    v8f acc = {0.f, 0.f, 0.f, 0.f, 0.f, 0.f, 0.f, 0.f};
    const int h8 = 8 * half;
    for (int k0 = 0; k0 < K; k0 += 32) {
        FragAB a, b;
#pragma unroll
        for (int v = 0; v < 8; ++v) {      // A: lane=row, 16 contiguous K
            int kb = 2 * v + 16 * half + k0;
            a.u[v] = arow[kb >> 1];
        }
        const u32x4* bp = (const u32x4*)(wrow + k0 + 16 * half);
        b.q[0] = bp[0]; b.q[1] = bp[1];    // B: lane=column, 16 contiguous K
        acc = __builtin_amdgcn_wmma_f32_16x16x32_bf16(false, a.v, false, b.v,
                                                      (short)0, acc, false, false);
    }
    const int   col = nt * 16 + l15;
    const float bv  = bias ? bias[col] : 0.0f;
#pragma unroll
    for (int r = 0; r < 8; ++r) {
        int row = mt * 16 + r + 8 * half;  // C/D layout: M = r + 8*half
        C[(size_t)row * N + col] = acc[r] + bv;
    }
}

// ---------------------------------------------------------------------------
// Row LayerNorm over H=512; one wave per row; optional fp32 and bf16 outputs.
__global__ void k_layernorm(const float* __restrict__ in,
                            float* __restrict__ outf,
                            unsigned short* __restrict__ outbf,
                            const float* __restrict__ g,
                            const float* __restrict__ beta,
                            int rows) {
    const int lane = threadIdx.x & 31;
    const int row  = blockIdx.x * (blockDim.x >> 5) + (threadIdx.x >> 5);
    if (row >= rows) return;
    const float* p = in + (size_t)row * HH;
    float v[16];
    float s = 0.f;
#pragma unroll
    for (int j = 0; j < 16; ++j) { v[j] = p[lane + 32 * j]; s += v[j]; }
#pragma unroll
    for (int o = 16; o >= 1; o >>= 1) s += __shfl_xor(s, o);
    const float mu = s * (1.0f / HH);
    float q = 0.f;
#pragma unroll
    for (int j = 0; j < 16; ++j) { float d = v[j] - mu; q += d * d; }
#pragma unroll
    for (int o = 16; o >= 1; o >>= 1) q += __shfl_xor(q, o);
    const float inv = rsqrtf(q * (1.0f / HH) + 1e-5f);
#pragma unroll
    for (int j = 0; j < 16; ++j) {
        int   c  = lane + 32 * j;
        float ov = (v[j] - mu) * inv * g[c] + beta[c];
        if (outf)  outf[(size_t)row * HH + c]  = ov;
        if (outbf) outbf[(size_t)row * HH + c] = f2bf(ov);
    }
}

// ---------------------------------------------------------------------------
// Distributed persistent LSTM layer across NWG=16 workgroups (one per WGP).
//
// WG g owns h-columns [32g, 32g+32): 8 gate tiles (i/f/g/o x 2 h-tiles), one
// per wave (256 threads = 8 waves). The (128 x 1024) bf16 weight shard is
// loaded into LDS once and reused for all T steps (ds_load_b128 B-fragments).
// Per step: A = [x_t | h_{t-1}] (16x1024 bf16) is DMA'd into LDS with
// global_load_async_to_lds_b128; gates go through an 8KB LDS staging tile so
// the cell update is thread-local (c in registers); the 1KB h-slice is
// published to a ping-pong L2 buffer and all WGs meet at an atomic grid
// barrier (monotonic ticket, s_sleep spin).
//
// Dynamic LDS layout (303104 B total, fits CDNA5's 320KB WGP budget):
//   [0,       262144)  W shard  (128 rows x 1024 bf16)
//   [262144,  294912)  A tile   (16 x 1024 bf16)
//   [294912,  303104)  gates    (4 x 16 x 32 fp32)
#define LSTM_LDS_W  0
#define LSTM_LDS_A  262144
#define LSTM_LDS_G  294912
#define LSTM_LDS_SZ 303104

__global__ __launch_bounds__(256)
void k_lstm_layer_mw(const unsigned short* __restrict__ Wl,   // (4H, 2H) bf16
                     const float* __restrict__ bl,            // (4H)
                     const unsigned short* __restrict__ xseq, // (B,T,H) bf16
                     unsigned short* __restrict__ hseq_bf,    // (B,T,H) bf16 or null
                     float* __restrict__ hseq_f32,            // (B,T,H) f32 or null
                     unsigned short* __restrict__ hbuf,       // ping-pong 2*(B,H) bf16, [0] pre-zeroed
                     unsigned int* __restrict__ barcnt) {     // pre-zeroed ticket counter
    extern __shared__ char smem[];
    unsigned short* Wsh = (unsigned short*)(smem + LSTM_LDS_W);
    unsigned short* Ash = (unsigned short*)(smem + LSTM_LDS_A);
    float*          Gsh = (float*)(smem + LSTM_LDS_G);

    const int g    = blockIdx.x;          // 0..15
    const int tid  = threadIdx.x;
    const int lane = tid & 31;
    const int w    = tid >> 5;            // 0..7
    const int half = lane >> 4;
    const int l15  = lane & 15;
    const int gI   = w & 3;               // gate (i/f/g/o)
    const int htl  = w >> 2;              // h-tile within WG (0..1)
    const int cbase = gI * HH + (2 * g + htl) * 16;   // global W row / gate col

    // ---- preload this wave's 32KB W tile into LDS (once per layer) ----
    for (int r = 0; r < 16; ++r) {
        const u32x4* src = (const u32x4*)(Wl + (size_t)(cbase + r) * (2 * HH));
        u32x4*       dst = (u32x4*)(Wsh + ((w * 16 + r) << 10));
        for (int ck = lane; ck < 128; ck += 32) dst[ck] = src[ck];
    }
    const float bv = bl[cbase + l15];
    __syncthreads();

    const unsigned int* arow = (const unsigned int*)(Ash + (l15 << 10));
    const int h8 = 8 * half;
    float cst0 = 0.f, cst1 = 0.f;         // cell state: 2 elements per thread

    for (int t = 0; t < TT; ++t) {
        const unsigned short* hread  = hbuf + (size_t)(t & 1) * BB * HH;
        unsigned short*       hwrite = hbuf + (size_t)((t + 1) & 1) * BB * HH;

        // ---- async-DMA A = [x_t | h_{t-1}] into LDS (2048 x 16B chunks) ----
        for (int u = tid; u < BB * 128; u += 256) {
            int b  = u >> 7;
            int cw = u & 127;             // 16B chunk within the 2KB row
            unsigned int ldsoff = LSTM_LDS_A + (b << 11) + (cw << 4);
            const void* src = (cw < 64)
                ? (const void*)(xseq + ((size_t)b * TT + t) * HH + (cw << 3))
                : (const void*)(hread + (size_t)b * HH + ((cw - 64) << 3));
            async_g2l_b128(ldsoff, src);
        }
        wait_asynccnt0();
        __syncthreads();

        // ---- one 16x16 gate tile per wave: 32 x WMMA, B from LDS ----
        v8f acc;
#pragma unroll
        for (int r = 0; r < 8; ++r) acc[r] = bv;
        for (int k0 = 0; k0 < 2 * HH; k0 += 32) {
            FragAB a, b;
#pragma unroll
            for (int v = 0; v < 8; ++v) {
                int kb = 2 * v + 16 * half + k0;
                a.u[v] = arow[kb >> 1];
            }
            const u32x4* bp = (const u32x4*)(Wsh + ((w * 16 + l15) << 10) + k0 + 16 * half);
            b.q[0] = bp[0]; b.q[1] = bp[1];
            acc = __builtin_amdgcn_wmma_f32_16x16x32_bf16(false, a.v, false, b.v,
                                                          (short)0, acc, false, false);
        }
        // stage gates: Gsh[gate][b][local col 0..31]
#pragma unroll
        for (int r = 0; r < 8; ++r) {
            Gsh[gI * (BB * 32) + (r + 8 * half) * 32 + htl * 16 + l15] = acc[r];
        }
        __syncthreads();

        // ---- wave-local cell update: 512 (b, col) pairs, 2 per thread ----
#pragma unroll
        for (int j = 0; j < 2; ++j) {
            int   p  = tid + 256 * j;
            int   b  = p >> 5;
            int   hl = p & 31;
            float iv = sigm(Gsh[0 * (BB * 32) + b * 32 + hl]);
            float fv = sigm(Gsh[1 * (BB * 32) + b * 32 + hl]);
            float gv = tanhf(Gsh[2 * (BB * 32) + b * 32 + hl]);
            float ov = sigm(Gsh[3 * (BB * 32) + b * 32 + hl]);
            float c  = fv * (j ? cst1 : cst0) + iv * gv;
            if (j) cst1 = c; else cst0 = c;
            float h  = ov * tanhf(c);
            int   col = 32 * g + hl;
            hwrite[(size_t)b * HH + col] = f2bf(h);
            if (hseq_bf)  hseq_bf[((size_t)b * TT + t) * HH + col]  = f2bf(h);
            if (hseq_f32) hseq_f32[((size_t)b * TT + t) * HH + col] = h;
        }

        // ---- device-scope grid barrier (monotonic ticket) ----
        __threadfence();
        __syncthreads();                  // Gsh reads done; h stores issued
        if (tid == 0) {
            __hip_atomic_fetch_add(barcnt, 1u, __ATOMIC_ACQ_REL, __HIP_MEMORY_SCOPE_AGENT);
            unsigned int target = (unsigned int)NWG * (unsigned int)(t + 1);
            while (__hip_atomic_load(barcnt, __ATOMIC_ACQUIRE, __HIP_MEMORY_SCOPE_AGENT) < target) {
                __builtin_amdgcn_s_sleep(1);
            }
        }
        __syncthreads();
    }
}

// ---------------------------------------------------------------------------
// Routing: logits (Hx3), softmax/tau, keep top-2 of 3 (drop the min; ties
// drop the highest index, matching top_k's lower-index preference), renorm.
__global__ void k_route(const float* __restrict__ y,
                        const float* __restrict__ Wm,   // (3,H)
                        const float* __restrict__ bm,   // (3)
                        float* __restrict__ mout) {     // (rows,3)
    const int lane = threadIdx.x & 31;
    const int row  = blockIdx.x * (blockDim.x >> 5) + (threadIdx.x >> 5);
    if (row >= ROWS) return;
    const float* p = y + (size_t)row * HH;
    float a0 = 0.f, a1 = 0.f, a2 = 0.f;
#pragma unroll
    for (int j = 0; j < 16; ++j) {
        int   c = lane + 32 * j;
        float v = p[c];
        a0 += v * Wm[c];
        a1 += v * Wm[HH + c];
        a2 += v * Wm[2 * HH + c];
    }
#pragma unroll
    for (int o = 16; o >= 1; o >>= 1) {
        a0 += __shfl_xor(a0, o);
        a1 += __shfl_xor(a1, o);
        a2 += __shfl_xor(a2, o);
    }
    if (lane == 0) {
        const float it = 1.0f / 0.7f;
        float l0 = (a0 + bm[0]) * it, l1 = (a1 + bm[1]) * it, l2 = (a2 + bm[2]) * it;
        float mx = fmaxf(l0, fmaxf(l1, l2));
        float e0 = __expf(l0 - mx), e1 = __expf(l1 - mx), e2 = __expf(l2 - mx);
        float Z  = e0 + e1 + e2;
        float p0 = e0 / Z, p1 = e1 / Z, p2 = e2 / Z;
        int   im = 0; float pm = p0;
        if (p1 <= pm) { pm = p1; im = 1; }
        if (p2 <= pm) { pm = p2; im = 2; }
        float s = fmaxf(1.0f - pm, 1e-8f);
        mout[(size_t)row * 3 + 0] = (im == 0) ? 0.f : p0 / s;
        mout[(size_t)row * 3 + 1] = (im == 1) ? 0.f : p1 / s;
        mout[(size_t)row * 3 + 2] = (im == 2) ? 0.f : p2 / s;
    }
}

// ---------------------------------------------------------------------------
// Head epilogue: ReLU -> LayerNorm -> scale by m[row, head].
__global__ void k_ff_relu_ln_scale(const float* __restrict__ z,
                                   const float* __restrict__ g,
                                   const float* __restrict__ beta,
                                   const float* __restrict__ mgate,  // (rows,3)
                                   int head,
                                   float* __restrict__ out) {
    const int lane = threadIdx.x & 31;
    const int row  = blockIdx.x * (blockDim.x >> 5) + (threadIdx.x >> 5);
    if (row >= ROWS) return;
    const float* p = z + (size_t)row * HH;
    float v[16];
    float s = 0.f;
#pragma unroll
    for (int j = 0; j < 16; ++j) { v[j] = fmaxf(p[lane + 32 * j], 0.0f); s += v[j]; }
#pragma unroll
    for (int o = 16; o >= 1; o >>= 1) s += __shfl_xor(s, o);
    const float mu = s * (1.0f / HH);
    float q = 0.f;
#pragma unroll
    for (int j = 0; j < 16; ++j) { float d = v[j] - mu; q += d * d; }
#pragma unroll
    for (int o = 16; o >= 1; o >>= 1) q += __shfl_xor(q, o);
    const float inv   = rsqrtf(q * (1.0f / HH) + 1e-5f);
    const float scale = mgate[(size_t)row * 3 + head];
#pragma unroll
    for (int j = 0; j < 16; ++j) {
        int c = lane + 32 * j;
        out[(size_t)row * HH + c] = ((v[j] - mu) * inv * g[c] + beta[c]) * scale;
    }
}

// ---------------------------------------------------------------------------
extern "C" void kernel_launch(void* const* d_in, const int* in_sizes, int n_in,
                              void* d_out, int out_size, void* d_ws, size_t ws_size,
                              hipStream_t stream) {
    const float* x       = (const float*)d_in[0];
    const float* pre_W   = (const float*)d_in[1];
    const float* pre_b   = (const float*)d_in[2];
    const float* pre_g   = (const float*)d_in[3];
    const float* pre_be  = (const float*)d_in[4];
    const float* lstm_W  = (const float*)d_in[5];
    const float* lstm_b  = (const float*)d_in[6];
    const float* post_g  = (const float*)d_in[7];
    const float* post_be = (const float*)d_in[8];
    const float* mask_W  = (const float*)d_in[9];
    const float* mask_b  = (const float*)d_in[10];
    const float* ff_W    = (const float*)d_in[11];
    const float* ff_b    = (const float*)d_in[12];
    const float* ff_g    = (const float*)d_in[13];
    const float* ff_be   = (const float*)d_in[14];

    // allow 296KB dynamic LDS for the LSTM kernel (CDNA5 WGP: 320KB)
    hipFuncSetAttribute((const void*)k_lstm_layer_mw,
                        hipFuncAttributeMaxDynamicSharedMemorySize, LSTM_LDS_SZ);

    // workspace carve (all offsets 256B-aligned; total ~142 MB)
    char* ws = (char*)d_ws;
    size_t off = 0;
    unsigned short* xbf    = (unsigned short*)(ws + off); off += (size_t)ROWS * DD * 2;
    unsigned short* preWb  = (unsigned short*)(ws + off); off += (size_t)HH * DD * 2;
    unsigned short* lstmWb = (unsigned short*)(ws + off); off += (size_t)NLAYER * 4 * HH * 2 * HH * 2;
    unsigned short* ffWb   = (unsigned short*)(ws + off); off += (size_t)NHEAD * HH * HH * 2;
    float*          yraw   = (float*)(ws + off);          off += (size_t)ROWS * HH * 4;
    unsigned short* seqA   = (unsigned short*)(ws + off); off += (size_t)ROWS * HH * 2;
    unsigned short* seqB   = (unsigned short*)(ws + off); off += (size_t)ROWS * HH * 2;
    float*          yseq   = (float*)(ws + off);          off += (size_t)ROWS * HH * 4;
    unsigned short* ybf    = (unsigned short*)(ws + off); off += (size_t)ROWS * HH * 2;
    unsigned short* hbuf   = (unsigned short*)(ws + off); off += (size_t)2 * BB * HH * 2;  // ping-pong h
    unsigned int*   barcnt = (unsigned int*)(ws + off);   off += 256;                      // grid barrier

    float* mout = (float*)d_out + (size_t)NHEAD * ROWS * HH;

    // fp32 -> bf16 weight/activation conversions
    { int n = ROWS * DD;                 k_f32_to_bf16<<<(n + 255) / 256, 256, 0, stream>>>(x,      xbf,    n); }
    { int n = HH * DD;                   k_f32_to_bf16<<<(n + 255) / 256, 256, 0, stream>>>(pre_W,  preWb,  n); }
    { int n = NLAYER * 4 * HH * 2 * HH;  k_f32_to_bf16<<<(n + 255) / 256, 256, 0, stream>>>(lstm_W, lstmWb, n); }
    { int n = NHEAD * HH * HH;           k_f32_to_bf16<<<(n + 255) / 256, 256, 0, stream>>>(ff_W,   ffWb,   n); }

    const int gemm_blocks = ((ROWS / 16) * (HH / 16)) / 8;  // 8 waves/block

    // pre-projection GEMM + pre LayerNorm (bf16 out feeds LSTM layer 0)
    k_wmma_gemm_bias<<<gemm_blocks, 256, 0, stream>>>(xbf, preWb, pre_b, yraw, ROWS, HH, DD);
    k_layernorm<<<ROWS / 8, 256, 0, stream>>>(yraw, nullptr, seqA, pre_g, pre_be, ROWS);

    // stacked LSTM: 16-WGP distributed persistent kernels.
    // zero h ping-pong (32KB) + barrier ticket (contiguous) before each layer.
    const int nz = (2 * BB * HH * 2) / 4 + 1;   // hbuf uints + barcnt
    k_zero32<<<(nz + 255) / 256, 256, 0, stream>>>((unsigned int*)hbuf, nz);
    k_lstm_layer_mw<<<NWG, 256, LSTM_LDS_SZ, stream>>>(
        lstmWb, lstm_b, seqA, seqB, nullptr, hbuf, barcnt);
    k_zero32<<<(nz + 255) / 256, 256, 0, stream>>>((unsigned int*)hbuf, nz);
    k_lstm_layer_mw<<<NWG, 256, LSTM_LDS_SZ, stream>>>(
        lstmWb + (size_t)4 * HH * 2 * HH, lstm_b + 4 * HH, seqB, nullptr, yseq, hbuf, barcnt);

    // post LayerNorm (fp32 for routing, bf16 for head GEMMs)
    k_layernorm<<<ROWS / 8, 256, 0, stream>>>(yseq, yraw, ybf, post_g, post_be, ROWS);

    // routing mask m -> tail of d_out
    k_route<<<ROWS / 8, 256, 0, stream>>>(yraw, mask_W, mask_b, mout);

    // per-head branch FF + ReLU/LN/gate -> d_out[n]
    for (int n = 0; n < NHEAD; ++n) {
        k_wmma_gemm_bias<<<gemm_blocks, 256, 0, stream>>>(
            ybf, ffWb + (size_t)n * HH * HH, ff_b + n * HH, yseq, ROWS, HH, HH);
        k_ff_relu_ln_scale<<<ROWS / 8, 256, 0, stream>>>(
            yseq, ff_g + n * HH, ff_be + n * HH, mout, n,
            (float*)d_out + (size_t)n * ROWS * HH);
    }
}
